// MaskedSlidingPSN_49134425866320
// MI455X (gfx1250) — matrix operand order repaired and checked by reading, not compiled, not run on hardware
//
#include <hip/hip_runtime.h>

// ---------------------------------------------------------------------------
// MaskedSlidingPSN (ORDER=32) as a banded bf16 WMMA GEMM + heaviside epilogue.
//   out[t,c] = step( sum_{k=0..31} weight[31-k] * x[t-k,c] + threshold )
// x: [T=1024, B*N=65536] f32, out same shape f32.
// Memory-bound: 256MB in + 256MB out => ~22us floor at 23.3 TB/s. Each wave
// reads its x column strip exactly once via a rolling 3-chunk window.
// ---------------------------------------------------------------------------

typedef __attribute__((ext_vector_type(16))) __bf16 v16bf;
typedef __attribute__((ext_vector_type(8)))  float  v8f;

#define T_DIM 1024
#define BN    65536      // B * N = 64 * 1024
#define ORDER 32
#define SEG   256        // rows of T handled per wave
#define NSEG  (T_DIM / SEG)          // 4
#define NSTRIP (BN / 16)             // 4096 column strips
#define WAVES_PER_BLOCK 8

// K index held by A element e for this half-wave (contiguous order,
// matching load_chunk): element e <-> K = 16*half + e.
__device__ __forceinline__ int a_kidx(int e, int half) {
  return half * 16 + e;
}

// Build the (wave-invariant) A pattern for a K-chunk at row offset `off`
// relative to the output tile: A(M,K) = weight[ORDER-1-d], d = M + off - K,
// zero outside the band. Powers-of-two weights are exact in bf16.
__device__ __forceinline__ v16bf make_a(const float* __restrict__ wgt,
                                        int off, int M, int half) {
  v16bf a;
#pragma unroll
  for (int e = 0; e < 16; ++e) {
    int K = a_kidx(e, half);
    int d = M + off - K;
    float w = (d >= 0 && d < ORDER) ? wgt[ORDER - 1 - d] : 0.0f;
    a[e] = (__bf16)w;
  }
  return a;
}

// Load one 32-row B chunk (rows [row0, row0+31], one column per lane group).
// B chunk in contiguous K order (same as a_kidx): lane half 0 holds K=0..15, half 1 holds K=16..31,
// element e <-> K = 16*half + e. Out-of-range T rows read as zero.
__device__ __forceinline__ v16bf load_chunk(const float* __restrict__ x,
                                            int row0, int col, int half) {
  v16bf b;
  const int base = row0 + half * 16;
#pragma unroll
  for (int e = 0; e < 16; ++e) {
    int row = base + e;
    float v = (row >= 0 && row < T_DIM) ? x[(long)row * BN + col] : 0.0f;
    b[e] = (__bf16)v;
  }
  return b;
}

// C/D f32 16x16 layout: VGPR r holds row M = r + 8*half, N = lane&15.
__device__ __forceinline__ void store_spikes(float* __restrict__ out,
                                             const v8f& acc, float thr,
                                             int t0, int col, int half) {
#pragma unroll
  for (int r = 0; r < 8; ++r) {
    int row = t0 + r + 8 * half;
    out[(long)row * BN + col] = ((acc[r] + thr) >= 0.0f) ? 1.0f : 0.0f;
  }
}

__global__ __launch_bounds__(256) void psn_wmma_kernel(
    const float* __restrict__ x, const float* __restrict__ weight,
    const float* __restrict__ thr_p, float* __restrict__ out) {
  const int lane  = threadIdx.x & 31;
  const int wave  = threadIdx.x >> 5;
  const int gwave = blockIdx.x * WAVES_PER_BLOCK + wave;
  const int seg   = gwave & (NSEG - 1);   // T segment (0..3)
  const int strip = gwave >> 2;           // column strip (0..4095)
  const int M     = lane & 15;
  const int half  = lane >> 4;
  const int col   = strip * 16 + M;
  const float thr = thr_p[0];
  const int t_begin = seg * SEG;

  // Five constant A-band patterns (offset of chunk start vs. tile start).
  const v16bf A0 = make_a(weight, 32, M, half);   // tile t0    vs chunk L
  const v16bf A1 = make_a(weight, 0, M, half);    // tile t0    vs chunk C
  const v16bf AL = make_a(weight, 48, M, half);   // tile t0+16 vs chunk L
  const v16bf AC = make_a(weight, 16, M, half);   // tile t0+16 vs chunk C
  const v16bf AN = make_a(weight, -16, M, half);  // tile t0+16 vs chunk N

  // Rolling window of three 32-row x chunks: L=[t0-32], C=[t0], N=[t0+32].
  v16bf Bl = load_chunk(x, t_begin - 32, col, half);
  v16bf Bc = load_chunk(x, t_begin, col, half);

  for (int t0 = t_begin; t0 < t_begin + SEG; t0 += 32) {
    v16bf Bn = load_chunk(x, t0 + 32, col, half);

    // Output tile rows [t0, t0+15]: band covers rows [t0-32, t0+15].
    v8f acc = {};
    acc = __builtin_amdgcn_wmma_f32_16x16x32_bf16(false, A0, false, Bl,
                                                  (short)0, acc, false, false);
    acc = __builtin_amdgcn_wmma_f32_16x16x32_bf16(false, A1, false, Bc,
                                                  (short)0, acc, false, false);
    store_spikes(out, acc, thr, t0, col, half);

    // Output tile rows [t0+16, t0+31]: band covers rows [t0-16, t0+31].
    v8f acc2 = {};
    acc2 = __builtin_amdgcn_wmma_f32_16x16x32_bf16(false, AL, false, Bl,
                                                   (short)0, acc2, false, false);
    acc2 = __builtin_amdgcn_wmma_f32_16x16x32_bf16(false, AC, false, Bc,
                                                   (short)0, acc2, false, false);
    acc2 = __builtin_amdgcn_wmma_f32_16x16x32_bf16(false, AN, false, Bn,
                                                   (short)0, acc2, false, false);
    store_spikes(out, acc2, thr, t0 + 16, col, half);

    Bl = Bc;  // slide the window: x is read exactly once per strip
    Bc = Bn;
  }
}

extern "C" void kernel_launch(void* const* d_in, const int* in_sizes, int n_in,
                              void* d_out, int out_size, void* d_ws,
                              size_t ws_size, hipStream_t stream) {
  (void)in_sizes; (void)n_in; (void)out_size; (void)d_ws; (void)ws_size;
  const float* x      = (const float*)d_in[0];   // [T, B, N] f32
  const float* weight = (const float*)d_in[1];   // [32] f32
  const float* thr    = (const float*)d_in[2];   // scalar f32
  float* out          = (float*)d_out;           // [T, B, N] f32

  dim3 block(WAVES_PER_BLOCK * 32);                         // 256 (8 waves)
  dim3 grid((NSTRIP * NSEG) / WAVES_PER_BLOCK);             // 2048 blocks
  psn_wmma_kernel<<<grid, block, 0, stream>>>(x, weight, thr, out);
}
